// HTR_VT_Swin_81011673137639
// MI455X (gfx1250) — matrix operand order repaired and checked by reading, not compiled, not run on hardware
//
#include <hip/hip_runtime.h>
#include <math.h>

// ---------------------------------------------------------------------------
// Swin block (shifted-window attention + MLP) for MI455X / gfx1250, wave32.
// All GEMMs use v_wmma_f32_16x16x32_f16 (f16 inputs, f32 accumulate).
// ---------------------------------------------------------------------------

#define T_TOK   262144      // B*H*W = 8*64*512
#define C_DIM   192
#define NHEADS  6
#define HD      32
#define IMG_H   64
#define IMG_W   512
#define WH      4
#define WWIN    8
#define SH      2
#define SW      4
#define NWIN_B  8192        // B * (H/WH) * (W/WWIN)
#define QKV_N   576
#define MLP_HID 768

typedef _Float16 v16h __attribute__((ext_vector_type(16)));
typedef _Float16 v8h  __attribute__((ext_vector_type(8)));
typedef float    v8f  __attribute__((ext_vector_type(8)));
typedef unsigned int u32x4 __attribute__((ext_vector_type(4)));

union H16 { v16h v; v8h h[2]; u32x4 u[2]; };

static __device__ __forceinline__ _Float16 u16_to_h(unsigned short s) {
    return __builtin_bit_cast(_Float16, s);
}

// ---------------------------------------------------------------------------
// f32 -> f16 conversion (weights)
// ---------------------------------------------------------------------------
__global__ void cvt_f32_f16(const float* __restrict__ in,
                            _Float16* __restrict__ out, int n) {
    int i = blockIdx.x * 256 + threadIdx.x;
    if (i < n) out[i] = (_Float16)in[i];
}

// ---------------------------------------------------------------------------
// LayerNorm over 192 channels, wave-per-token, f16 output.
// ---------------------------------------------------------------------------
__global__ __launch_bounds__(256) void ln_f16(const float* __restrict__ x,
                                              const float* __restrict__ g,
                                              const float* __restrict__ b,
                                              _Float16* __restrict__ out) {
    int lane = threadIdx.x & 31;
    int wave = threadIdx.x >> 5;
    size_t t = (size_t)blockIdx.x * 8 + wave;
    const float* row = x + t * C_DIM;
    int c0 = lane * 6;
    float v[6];
#pragma unroll
    for (int i = 0; i < 6; ++i) v[i] = row[c0 + i];
    float s = 0.f, ss = 0.f;
#pragma unroll
    for (int i = 0; i < 6; ++i) { s += v[i]; ss += v[i] * v[i]; }
#pragma unroll
    for (int o = 16; o >= 1; o >>= 1) {
        s  += __shfl_xor(s, o, 32);
        ss += __shfl_xor(ss, o, 32);
    }
    float mu  = s * (1.f / 192.f);
    float var = ss * (1.f / 192.f) - mu * mu;
    float rs  = rsqrtf(var + 1e-5f);
    _Float16 o6[6];
#pragma unroll
    for (int i = 0; i < 6; ++i)
        o6[i] = (_Float16)(((v[i] - mu) * rs) * g[c0 + i] + b[c0 + i]);
    unsigned* dst = (unsigned*)(out + t * C_DIM + c0);
#pragma unroll
    for (int i = 0; i < 3; ++i) {
        unsigned lo = __builtin_bit_cast(unsigned short, o6[2 * i]);
        unsigned hi = __builtin_bit_cast(unsigned short, o6[2 * i + 1]);
        dst[i] = lo | (hi << 16);
    }
}

// ---------------------------------------------------------------------------
// Generic WMMA GEMM: out = A[M,K] * B[K,N] + bias (+ residual / GELU).
// Block tile 256x64, 8 waves; each wave owns 32 rows x 64 cols
// (2x4 WMMA tiles => 8 WMMAs per 32-K step). Double-buffered LDS,
// one barrier per K-step.
// ---------------------------------------------------------------------------
#define PADK 40
#define EPI_H16      0
#define EPI_RES_F32  1
#define EPI_GELU_H16 2

__global__ __launch_bounds__(256) void gemm_wmma(
    const _Float16* __restrict__ A, const _Float16* __restrict__ B,
    const float* __restrict__ bias, const float* __restrict__ resid,
    _Float16* __restrict__ outH, float* __restrict__ outF,
    int M, int N, int K, int epi) {
    __shared__ _Float16 sA[2][256 * PADK];   // rows x 32 halves (padded)
    __shared__ _Float16 sB[2][64 * PADK];    // [n][k] transposed (padded)

    int tid  = threadIdx.x;
    int lane = tid & 31;
    int wave = tid >> 5;          // 0..7: 32-row slab within the 256-row tile
    int n0 = blockIdx.x * 64;
    int m0 = blockIdx.y * 256;
    int l15 = lane & 15;
    int lh  = lane >> 4;
    int kk   = tid >> 3;          // 0..31 : k-row for B staging
    int nseg = tid & 7;           // 0..7  : 8-col segment for B staging

    v8f acc[2][4] = {};
    u32x4 ra[4];                  // staged A row (64B = 32 halves)
    u32x4 rb;                     // staged B chunk (8 halves of one k-row)

    const int kSteps = K >> 5;

    auto globalFetch = [&](int k0) {
        const u32x4* srcA = (const u32x4*)(A + (size_t)(m0 + tid) * K + k0);
#pragma unroll
        for (int c = 0; c < 4; ++c) ra[c] = srcA[c];
        rb = *(const u32x4*)(B + (size_t)(k0 + kk) * N + n0 + nseg * 8);
    };
    auto ldsStore = [&](int buf) {
        u32x4* dstA = (u32x4*)(sA[buf] + tid * PADK);
#pragma unroll
        for (int c = 0; c < 4; ++c) dstA[c] = ra[c];
        int nb = nseg * 8;
#pragma unroll
        for (int e = 0; e < 4; ++e) {
            unsigned w = rb[e];
            sB[buf][(nb + 2 * e + 0) * PADK + kk] =
                u16_to_h((unsigned short)(w & 0xffff));
            sB[buf][(nb + 2 * e + 1) * PADK + kk] =
                u16_to_h((unsigned short)(w >> 16));
        }
    };

    globalFetch(0);
    ldsStore(0);
    __syncthreads();

    for (int step = 0; step < kSteps; ++step) {
        int cur = step & 1;
        bool hasNext = (step + 1) < kSteps;
        if (hasNext) globalFetch((step + 1) << 5);

        H16 af[2], bf[4];
#pragma unroll
        for (int mt = 0; mt < 2; ++mt) {
            const _Float16* base =
                sA[cur] + (wave * 32 + mt * 16 + l15) * PADK + lh * 8;
            af[mt].u[0] = *(const u32x4*)(base);
            af[mt].u[1] = *(const u32x4*)(base + 16);
        }
#pragma unroll
        for (int nt = 0; nt < 4; ++nt) {
            const _Float16* base = sB[cur] + (nt * 16 + l15) * PADK + lh * 8;
            bf[nt].u[0] = *(const u32x4*)(base);
            bf[nt].u[1] = *(const u32x4*)(base + 16);
        }
#pragma unroll
        for (int mt = 0; mt < 2; ++mt)
#pragma unroll
            for (int nt = 0; nt < 4; ++nt)
                acc[mt][nt] = __builtin_amdgcn_wmma_f32_16x16x32_f16(
                    false, af[mt].v, false, bf[nt].v, (short)0, acc[mt][nt],
                    false, false);

        if (hasNext) ldsStore(1 - cur);
        __syncthreads();
    }

#pragma unroll
    for (int mt = 0; mt < 2; ++mt) {
#pragma unroll
        for (int nt = 0; nt < 4; ++nt) {
            int col = n0 + nt * 16 + l15;
            float bv = bias[col];
#pragma unroll
            for (int r = 0; r < 8; ++r) {
                int rowg = m0 + wave * 32 + mt * 16 + r + lh * 8;
                size_t idx = (size_t)rowg * N + col;
                float v = acc[mt][nt][r] + bv;
                if (epi == EPI_RES_F32) {
                    outF[idx] = v + resid[idx];
                } else if (epi == EPI_GELU_H16) {
                    float gl = 0.5f * v * (1.0f + erff(v * 0.70710678118654752f));
                    outH[idx] = (_Float16)gl;
                } else {
                    outH[idx] = (_Float16)v;
                }
            }
        }
    }
}

// ---------------------------------------------------------------------------
// Shifted-window attention: one wave per (window, head).
// qkv: f16 [T, 576] in natural token order, cols = {q,k,v} x head x 32.
// attn_out: f16 [T, 192] in natural token order (window-reverse+unshift fused).
// ---------------------------------------------------------------------------
__global__ __launch_bounds__(256) void swin_attn(
    const _Float16* __restrict__ qkv, const float* __restrict__ bias_table,
    _Float16* __restrict__ attn_out) {
    __shared__ float    sS[8][32][36];   // scores (f32), later reused for P (f16)
    __shared__ _Float16 sVt[8][32][40];  // V transposed: [d][m]

    int lane = threadIdx.x & 31;
    int wave = threadIdx.x >> 5;
    int l15 = lane & 15;
    int lh  = lane >> 4;

    int task = blockIdx.x * 8 + wave;       // 0 .. 49151
    int head = task % NHEADS;
    int win  = task / NHEADS;               // 0 .. 8191
    int b    = win >> 10;
    int wrem = win & 1023;
    int wi = wrem >> 6;                     // 0..15
    int wj = wrem & 63;                     // 0..63

    // in-window index n (0..31) -> natural-order token index (un-roll applied)
    auto tok = [&](int n) -> size_t {
        int i = n >> 3, j = n & 7;
        int h = wi * WH + i + SH;   if (h >= IMG_H) h -= IMG_H;
        int w = wj * WWIN + j + SW; if (w >= IMG_W) w -= IMG_W;
        return (size_t)b * (IMG_H * IMG_W) + (size_t)h * IMG_W + w;
    };

    // ---- stage V transposed into LDS (lane = token m) ----
    {
        size_t tm = tok(lane);
        const u32x4* vsrc = (const u32x4*)(qkv + tm * QKV_N + 384 + head * HD);
#pragma unroll
        for (int c = 0; c < 4; ++c) {
            u32x4 vv = vsrc[c];
#pragma unroll
            for (int e = 0; e < 4; ++e) {
                unsigned w = vv[e];
                int d = c * 8 + e * 2;
                sVt[wave][d][lane]     = u16_to_h((unsigned short)(w & 0xffff));
                sVt[wave][d + 1][lane] = u16_to_h((unsigned short)(w >> 16));
            }
        }
    }

    // ---- load Q (A-fragments) and K (B-fragments) straight from global ----
    H16 qf[2], kf[2];
#pragma unroll
    for (int mt = 0; mt < 2; ++mt) {
        size_t tq = tok(mt * 16 + l15);
        const _Float16* base = qkv + tq * QKV_N + head * HD + lh * 8;
        qf[mt].u[0] = *(const u32x4*)(base);
        qf[mt].u[1] = *(const u32x4*)(base + 16);
    }
#pragma unroll
    for (int nt = 0; nt < 2; ++nt) {
        size_t tk = tok(nt * 16 + l15);
        const _Float16* base = qkv + tk * QKV_N + 192 + head * HD + lh * 8;
        kf[nt].u[0] = *(const u32x4*)(base);
        kf[nt].u[1] = *(const u32x4*)(base + 16);
    }

    // ---- S = q @ k^T  (4 WMMAs) ----
    v8f sacc[2][2] = {};
#pragma unroll
    for (int mt = 0; mt < 2; ++mt)
#pragma unroll
        for (int nt = 0; nt < 2; ++nt)
            sacc[mt][nt] = __builtin_amdgcn_wmma_f32_16x16x32_f16(
                false, qf[mt].v, false, kf[nt].v, (short)0, sacc[mt][nt],
                false, false);

#pragma unroll
    for (int mt = 0; mt < 2; ++mt)
#pragma unroll
        for (int nt = 0; nt < 2; ++nt)
#pragma unroll
            for (int r = 0; r < 8; ++r)
                sS[wave][mt * 16 + r + lh * 8][nt * 16 + l15] = sacc[mt][nt][r];
    __syncthreads();

    // ---- softmax: lane handles query row n = lane ----
    {
        const float scale = 0.17677669529663687f;  // 1/sqrt(32)
        int iq = lane >> 3, jq = lane & 7;
        int hsq = wi * WH + iq, wsq = wj * WWIN + jq;
        int idq = ((hsq >= IMG_H - SH) ? 2 : 0) + ((wsq >= IMG_W - SW) ? 1 : 0);
        float rowv[32];
        float mx = -1e30f;
#pragma unroll
        for (int m = 0; m < 32; ++m) {
            int im = m >> 3, jm = m & 7;
            int hsm = wi * WH + im, wsm = wj * WWIN + jm;
            int idm = ((hsm >= IMG_H - SH) ? 2 : 0) + ((wsm >= IMG_W - SW) ? 1 : 0);
            int dh = iq - im + (WH - 1);
            int dw = jq - jm + (WWIN - 1);
            float bias = bias_table[(dh * 15 + dw) * NHEADS + head];
            float val = sS[wave][lane][m] * scale + bias +
                        ((idm != idq) ? -100.0f : 0.0f);
            rowv[m] = val;
            mx = fmaxf(mx, val);
        }
        float sum = 0.f;
#pragma unroll
        for (int m = 0; m < 32; ++m) { rowv[m] = __expf(rowv[m] - mx); sum += rowv[m]; }
        float inv = 1.f / sum;
        _Float16* prow = (_Float16*)&sS[wave][lane][0];  // reuse S row as f16 P row
#pragma unroll
        for (int m = 0; m < 32; ++m) prow[m] = (_Float16)(rowv[m] * inv);
    }
    __syncthreads();

    // ---- O = P @ V  (4 WMMAs); P from sS (f16), V from sVt ----
    H16 pf[2], vf[2];
#pragma unroll
    for (int mt = 0; mt < 2; ++mt) {
        const _Float16* base =
            (const _Float16*)&sS[wave][mt * 16 + l15][0] + lh * 8;
        pf[mt].u[0] = *(const u32x4*)(base);
        pf[mt].u[1] = *(const u32x4*)(base + 16);
    }
#pragma unroll
    for (int nt = 0; nt < 2; ++nt) {
        const _Float16* base = &sVt[wave][nt * 16 + l15][0] + lh * 8;
        vf[nt].u[0] = *(const u32x4*)(base);
        vf[nt].u[1] = *(const u32x4*)(base + 16);
    }
    v8f oacc[2][2] = {};
#pragma unroll
    for (int mt = 0; mt < 2; ++mt)
#pragma unroll
        for (int nt = 0; nt < 2; ++nt)
            oacc[mt][nt] = __builtin_amdgcn_wmma_f32_16x16x32_f16(
                false, pf[mt].v, false, vf[nt].v, (short)0, oacc[mt][nt],
                false, false);

    // ---- scatter to natural token order (window-reverse + unshift) ----
#pragma unroll
    for (int mt = 0; mt < 2; ++mt) {
#pragma unroll
        for (int nt = 0; nt < 2; ++nt) {
#pragma unroll
            for (int r = 0; r < 8; ++r) {
                int n = mt * 16 + r + lh * 8;
                size_t t = tok(n);
                int col = head * HD + nt * 16 + l15;
                attn_out[t * C_DIM + col] = (_Float16)oacc[mt][nt][r];
            }
        }
    }
}

// ---------------------------------------------------------------------------
// Launch
// ---------------------------------------------------------------------------
extern "C" void kernel_launch(void* const* d_in, const int* in_sizes, int n_in,
                              void* d_out, int out_size, void* d_ws,
                              size_t ws_size, hipStream_t stream) {
    const float* x    = (const float*)d_in[0];
    const float* g1   = (const float*)d_in[3];
    const float* be1  = (const float*)d_in[4];
    const float* wqkv = (const float*)d_in[5];
    const float* bqkv = (const float*)d_in[6];
    const float* wpr  = (const float*)d_in[7];
    const float* bpr  = (const float*)d_in[8];
    const float* btab = (const float*)d_in[9];
    const float* g2   = (const float*)d_in[10];
    const float* be2  = (const float*)d_in[11];
    const float* w1   = (const float*)d_in[12];
    const float* bm1  = (const float*)d_in[13];
    const float* w2   = (const float*)d_in[14];
    const float* bm2  = (const float*)d_in[15];
    float* out = (float*)d_out;

    char* ws = (char*)d_ws;
    size_t off = 0;
    auto take = [&](size_t bytes) -> char* {
        char* p = ws + off;
        off = (off + bytes + 255) & ~(size_t)255;
        return p;
    };
    _Float16* wqkv_h = (_Float16*)take((size_t)192 * 576 * 2);
    _Float16* wpr_h  = (_Float16*)take((size_t)192 * 192 * 2);
    _Float16* w1_h   = (_Float16*)take((size_t)192 * 768 * 2);
    _Float16* w2_h   = (_Float16*)take((size_t)768 * 192 * 2);
    _Float16* xn_h   = (_Float16*)take((size_t)T_TOK * 192 * 2);  // LN1 out, then LN2 out
    _Float16* qh_h   = (_Float16*)take((size_t)T_TOK * 768 * 2);  // qkv, later MLP hidden
    _Float16* attn_h = (_Float16*)take((size_t)T_TOK * 192 * 2);
    float*    x1     = (float*)take((size_t)T_TOK * 192 * 4);

    cvt_f32_f16<<<(192 * 576 + 255) / 256, 256, 0, stream>>>(wqkv, wqkv_h, 192 * 576);
    cvt_f32_f16<<<(192 * 192 + 255) / 256, 256, 0, stream>>>(wpr, wpr_h, 192 * 192);
    cvt_f32_f16<<<(192 * 768 + 255) / 256, 256, 0, stream>>>(w1, w1_h, 192 * 768);
    cvt_f32_f16<<<(768 * 192 + 255) / 256, 256, 0, stream>>>(w2, w2_h, 768 * 192);

    // LN1(x) -> xn_h
    ln_f16<<<T_TOK / 8, 256, 0, stream>>>(x, g1, be1, xn_h);

    // qkv = LN1(x) @ w_qkv + b_qkv   (f16 out)
    gemm_wmma<<<dim3(QKV_N / 64, T_TOK / 256), 256, 0, stream>>>(
        xn_h, wqkv_h, bqkv, nullptr, qh_h, nullptr, T_TOK, QKV_N, 192, EPI_H16);

    // windowed attention (gather/scatter with shift fused)
    swin_attn<<<(NWIN_B * NHEADS) / 8, 256, 0, stream>>>(qh_h, btab, attn_h);

    // x1 = x + attn @ w_proj + b_proj   (f32 out)
    gemm_wmma<<<dim3(C_DIM / 64, T_TOK / 256), 256, 0, stream>>>(
        attn_h, wpr_h, bpr, x, nullptr, x1, T_TOK, C_DIM, 192, EPI_RES_F32);

    // LN2(x1) -> xn_h (reuse)
    ln_f16<<<T_TOK / 8, 256, 0, stream>>>(x1, g2, be2, xn_h);

    // h = gelu(LN2(x1) @ w1 + bm1)   (f16 out, reuse qh_h)
    gemm_wmma<<<dim3(MLP_HID / 64, T_TOK / 256), 256, 0, stream>>>(
        xn_h, w1_h, bm1, nullptr, qh_h, nullptr, T_TOK, MLP_HID, 192, EPI_GELU_H16);

    // out = x1 + h @ w2 + bm2   (f32 out)
    gemm_wmma<<<dim3(C_DIM / 64, T_TOK / 256), 256, 0, stream>>>(
        qh_h, w2_h, bm2, x1, nullptr, out, T_TOK, C_DIM, MLP_HID, EPI_RES_F32);
}
